// PosUpdateLayer_16020228014618
// MI455X (gfx1250) — matrix operand order repaired and checked by reading, not compiled, not run on hardware
//
#include <hip/hip_runtime.h>
#include <hip/hip_bf16.h>

typedef __attribute__((ext_vector_type(16))) _Float16 v16h;
typedef __attribute__((ext_vector_type(8)))  _Float16 v8h;
typedef __attribute__((ext_vector_type(8)))  float    v8f;

#define NHEADS 16
#define DHD    8
#define HID    128
#define DIM    128
#define EFEAT  16
#define KVIN   272
#define KVPAD  288

// ---------------- helpers ----------------

__device__ __forceinline__ void atomicMaxF(float* addr, float val) {
  if (val >= 0.0f) atomicMax((int*)addr, __float_as_int(val));
  else             atomicMin((unsigned int*)addr, __float_as_uint(val));
}

// A fragment: lane L holds row (L&15); halves 0..7 -> K = kb + (L<16?0:8) + j,
// halves 8..15 -> K = kb + 16 + (L<16?0:8) + (j-8)   (ISA 16-bit A 16x32 layout)
__device__ __forceinline__ v16h load_afrag(const _Float16* A, int strideH, int lane, int kb) {
  const int row = lane & 15;
  const int lo  = (lane < 16) ? 0 : 8;
  const _Float16* p = A + row * strideH + kb + lo;
  v16h a;
  ((uint4*)&a)[0] = *(const uint4*)(p);
  ((uint4*)&a)[1] = *(const uint4*)(p + 16);
  return a;
}

// B fragment comes pre-swizzled from the pack kernel: 16 contiguous halves per lane.
__device__ __forceinline__ v16h load_bfrag(const _Float16* P, int KC, int nt, int kc, int lane) {
  const uint4* bp = (const uint4*)(P + ((size_t)((nt * KC + kc) * 32 + lane) << 4));
  v16h b;
  ((uint4*)&b)[0] = bp[0];
  ((uint4*)&b)[1] = bp[1];
  return b;
}

template<int KC, int NT>
__device__ __forceinline__ void gemm_tile(const _Float16* A, int As, const _Float16* P,
                                          int lane, v8f* acc) {
  const v8f zero = {};
  #pragma unroll
  for (int nt = 0; nt < NT; ++nt) acc[nt] = zero;
  #pragma unroll
  for (int kc = 0; kc < KC; ++kc) {
    v16h a = load_afrag(A, As, lane, kc * 32);
    #pragma unroll
    for (int nt = 0; nt < NT; ++nt) {
      v16h b = load_bfrag(P, KC, nt, kc, lane);
      acc[nt] = __builtin_amdgcn_wmma_f32_16x16x32_f16(false, a, false, b,
                                                       (short)0, acc[nt], false, false);
    }
  }
}

// Bias + LayerNorm(128) + ReLU on an 8x(16x16) accumulator set, emit f16 A-matrix to LDS.
// C layout: acc[nt][r] = element (row=(lane<16 ? r : r+8), col=nt*16+(lane&15)).
__device__ __forceinline__ void ln_relu_to_lds(v8f* acc, const float* __restrict__ b1,
                                               const float* __restrict__ g,
                                               const float* __restrict__ be,
                                               _Float16* A2, int A2s, int lane) {
  const int c15 = lane & 15;
  float bias[8], gg[8], bb[8];
  #pragma unroll
  for (int nt = 0; nt < 8; ++nt) {
    int col = nt * 16 + c15;
    bias[nt] = b1[col]; gg[nt] = g[col]; bb[nt] = be[col];
  }
  #pragma unroll
  for (int nt = 0; nt < 8; ++nt)
    #pragma unroll
    for (int r = 0; r < 8; ++r)
      acc[nt][r] = acc[nt][r] + bias[nt];

  #pragma unroll
  for (int r = 0; r < 8; ++r) {
    float s = 0.f, s2 = 0.f;
    #pragma unroll
    for (int nt = 0; nt < 8; ++nt) { float v = acc[nt][r]; s += v; s2 += v * v; }
    // row is striped over 16 lanes within each half-wave: xor masks 1,2,4,8
    #pragma unroll
    for (int m = 1; m < 16; m <<= 1) { s += __shfl_xor(s, m); s2 += __shfl_xor(s2, m); }
    float mean = s * (1.0f / 128.0f);
    float var  = s2 * (1.0f / 128.0f) - mean * mean;
    float rstd = rsqrtf(var + 1e-5f);
    int R = (lane < 16) ? r : (r + 8);
    #pragma unroll
    for (int nt = 0; nt < 8; ++nt) {
      float v = (acc[nt][r] - mean) * rstd * gg[nt] + bb[nt];
      A2[R * A2s + nt * 16 + c15] = (_Float16)fmaxf(v, 0.0f);
    }
  }
}

// ---------------- kernels ----------------

__global__ void init_kernel(float* __restrict__ smax, float* __restrict__ denom,
                            float* __restrict__ out, int n16, int n3) {
  int i = blockIdx.x * blockDim.x + threadIdx.x;
  int stride = gridDim.x * blockDim.x;
  for (int j = i; j < n16; j += stride) { smax[j] = -__builtin_inff(); denom[j] = 0.0f; }
  for (int j = i; j < n3; j += stride) out[j] = 0.0f;
}

// f32 -> f16 streaming conversion (for node features h)
__global__ void conv_h_kernel(const float* __restrict__ src, _Float16* __restrict__ dst, int total4) {
  // total4 = total/4, processes 4 elems per thread-iteration
  for (int i = blockIdx.x * blockDim.x + threadIdx.x; i < total4; i += gridDim.x * blockDim.x) {
    float4 f = *(const float4*)&src[(size_t)i * 4];
    _Float16* p = &dst[(size_t)i * 4];
    p[0] = (_Float16)f.x; p[1] = (_Float16)f.y; p[2] = (_Float16)f.z; p[3] = (_Float16)f.w;
  }
}

// W[K][Nout] f32 -> packed f16 B fragments: idx = ((nt*KC+kc)*32+lane)*16+j
// n = nt*16 + (lane&15); k = kc*32 + (j<8 ? (lane<16?0:8) + j : 16 + (lane<16?0:8) + (j-8)); zero pad k>=K.
__global__ void pack_w_kernel(const float* __restrict__ W, _Float16* __restrict__ P,
                              int K, int KC, int Nout) {
  int NT = Nout >> 4;
  int total = KC * NT * 512;
  for (int i = blockIdx.x * blockDim.x + threadIdx.x; i < total; i += gridDim.x * blockDim.x) {
    int j    = i & 15;
    int lane = (i >> 4) & 31;
    int blk  = i >> 9;            // nt*KC + kc
    int kc   = blk % KC;
    int nt   = blk / KC;
    int n = nt * 16 + (lane & 15);
    int k = kc * 32 + ((lane < 16) ? 0 : 8) + ((j < 8) ? j : 8 + j);
    float v = (k < K) ? W[(size_t)k * Nout + n] : 0.0f;
    P[i] = (_Float16)v;
  }
}

// q = MLP(h): [N,128] -> LN -> ReLU -> [N,128], stored as f16
__global__ __launch_bounds__(128) void qmlp_kernel(
    const _Float16* __restrict__ h16,
    const _Float16* __restrict__ Pw1, const float* __restrict__ b1,
    const float* __restrict__ g, const float* __restrict__ be,
    const _Float16* __restrict__ Pw2, const float* __restrict__ b2,
    _Float16* __restrict__ qout, int N, int Ntiles) {
  const int AS = 136;
  __shared__ __align__(16) _Float16 A1[4][16 * 136];
  __shared__ __align__(16) _Float16 A2[4][16 * 136];
  const int wave = threadIdx.x >> 5;
  const int lane = threadIdx.x & 31;
  int tile = blockIdx.x * 4 + wave;
  if (tile >= Ntiles) tile = Ntiles - 1;
  const int n0 = tile * 16;
  _Float16* a1 = A1[wave];
  _Float16* a2 = A2[wave];

  // stage: 16 rows x 128 halves, 8-half (16B) chunks -> 256 chunks / 32 lanes
  #pragma unroll
  for (int it = 0; it < 8; ++it) {
    int c   = it * 32 + lane;      // chunk id
    int row = c >> 4;              // 16 chunks per row
    int j   = (c & 15) * 8;
    int node = n0 + row; if (node > N - 1) node = N - 1;
    *(uint4*)&a1[row * AS + j] = *(const uint4*)&h16[(size_t)node * DIM + j];
  }
  __syncthreads();

  v8f acc[8];
  gemm_tile<4, 8>(a1, AS, Pw1, lane, acc);
  ln_relu_to_lds(acc, b1, g, be, a2, AS, lane);
  __syncthreads();
  gemm_tile<4, 8>(a2, AS, Pw2, lane, acc);

  const int c15 = lane & 15;
  float bout[8];
  #pragma unroll
  for (int nt = 0; nt < 8; ++nt) bout[nt] = b2[nt * 16 + c15];
  #pragma unroll
  for (int nt = 0; nt < 8; ++nt)
    #pragma unroll
    for (int r = 0; r < 8; ++r) {
      int R = (lane < 16) ? r : r + 8;
      int node = n0 + R; if (node > N - 1) node = N - 1;
      qout[(size_t)node * DIM + nt * 16 + c15] = (_Float16)(acc[nt][r] + bout[nt]);
    }
}

// Per-edge k/v MLPs + scores + segment-max. One 16-edge tile per wave.
__global__ __launch_bounds__(128) void edge_kernel(
    const _Float16* __restrict__ h16, const float* __restrict__ edge_feat,
    const float* __restrict__ e_w, const long long* __restrict__ ei,
    const _Float16* __restrict__ q16,
    const _Float16* __restrict__ Pk1, const float* __restrict__ kb1,
    const float* __restrict__ kg, const float* __restrict__ kbe,
    const _Float16* __restrict__ Pk2, const float* __restrict__ kb2,
    const _Float16* __restrict__ Pv1, const float* __restrict__ vb1,
    const float* __restrict__ vg, const float* __restrict__ vbe,
    const _Float16* __restrict__ Pv2, const float* __restrict__ vb2,
    float* __restrict__ scores, float* __restrict__ vsc, float* __restrict__ smax,
    int E, int Etiles) {
  const int AS1 = 296;   // 288 halves + pad
  const int AS2 = 144;   // 128 halves + pad
  __shared__ __align__(16) _Float16 A1[4][16 * 296];   // later reused as float k[16][128]
  __shared__ __align__(16) _Float16 A2[4][16 * 144];
  __shared__ int idxs[4][32];                          // [0..15]=dst, [16..31]=src

  const int wave = threadIdx.x >> 5;
  const int lane = threadIdx.x & 31;
  int tile = blockIdx.x * 4 + wave;
  if (tile >= Etiles) tile = Etiles - 1;
  const int e0 = tile * 16;
  _Float16* a1 = A1[wave];
  _Float16* a2 = A2[wave];

  { int e = e0 + (lane & 15); if (e > E - 1) e = E - 1;
    idxs[wave][lane] = (int)((lane < 16) ? ei[(size_t)E + e] : ei[e]); }
  __syncthreads();

  // ---- stage A1 = [edge_feat(16) | h16[dst](128) | h16[src](128) | 0(16)] ----
  // edge_feat: 16 rows x 16 f32 = 64 quads -> 2 iterations (with f32->f16 convert)
  #pragma unroll
  for (int it = 0; it < 2; ++it) {
    int c   = it * 32 + lane;      // quad id, 4 per row
    int row = c >> 2;
    int c4  = (c & 3) * 4;
    int e = e0 + row; if (e > E - 1) e = E - 1;
    float4 f = *(const float4*)&edge_feat[(size_t)e * EFEAT + c4];
    _Float16* p = &a1[row * AS1 + c4];
    p[0] = (_Float16)f.x; p[1] = (_Float16)f.y; p[2] = (_Float16)f.z; p[3] = (_Float16)f.w;
  }
  // h16[dst] | h16[src]: 16 rows x 256 halves, 8-half chunks -> 512 chunks / 32 lanes
  #pragma unroll
  for (int it = 0; it < 16; ++it) {
    int c   = it * 32 + lane;
    int row = c >> 5;              // 32 chunks per row
    int j   = (c & 31) * 8;        // half-offset within the 256-half span
    int halfcol = EFEAT + j;       // 16..271
    const _Float16* src;
    if (halfcol < EFEAT + DIM) src = &h16[(size_t)idxs[wave][row] * DIM + (halfcol - EFEAT)];
    else                       src = &h16[(size_t)idxs[wave][16 + row] * DIM + (halfcol - EFEAT - DIM)];
    *(uint4*)&a1[row * AS1 + halfcol] = *(const uint4*)src;
  }
  // zero pad cols 272..287: 16 rows x 2 chunks of 8 halves = 32 chunks -> 1 iteration
  {
    uint4 z; z.x = 0u; z.y = 0u; z.z = 0u; z.w = 0u;
    int row = lane >> 1;
    int j   = KVIN + (lane & 1) * 8;
    *(uint4*)&a1[row * AS1 + j] = z;
  }
  __syncthreads();

  const int c15 = lane & 15;
  v8f acc[8];

  // ---- value MLP first (so A1 stays intact for the key MLP) ----
  gemm_tile<9, 8>(a1, AS1, Pv1, lane, acc);
  ln_relu_to_lds(acc, vb1, vg, vbe, a2, AS2, lane);
  __syncthreads();
  gemm_tile<4, 1>(a2, AS2, Pv2, lane, acc);
  {
    float bv = vb2[c15];
    #pragma unroll
    for (int r = 0; r < 8; ++r) {
      int R = (lane < 16) ? r : r + 8;
      int e = e0 + R; if (e > E - 1) e = E - 1;
      vsc[(size_t)e * NHEADS + c15] = (acc[0][r] + bv) * e_w[e];
    }
  }
  __syncthreads();

  // ---- key MLP ----
  gemm_tile<9, 8>(a1, AS1, Pk1, lane, acc);
  ln_relu_to_lds(acc, kb1, kg, kbe, a2, AS2, lane);
  __syncthreads();
  gemm_tile<4, 8>(a2, AS2, Pk2, lane, acc);

  float* kf = (float*)a1;   // reuse A1 region: k tile as f32 [16][128]
  {
    float bk[8];
    #pragma unroll
    for (int nt = 0; nt < 8; ++nt) bk[nt] = kb2[nt * 16 + c15];
    #pragma unroll
    for (int nt = 0; nt < 8; ++nt)
      #pragma unroll
      for (int r = 0; r < 8; ++r) {
        int R = (lane < 16) ? r : r + 8;
        kf[R * HID + nt * 16 + c15] = acc[nt][r] + bk[nt];
      }
  }
  __syncthreads();

  // ---- scores + segment max:  (e,h) pairs over 8 iterations x 32 lanes ----
  const float inv_sqrt_dh = 0.35355339059327373f;
  #pragma unroll
  for (int it = 0; it < 8; ++it) {
    int hh = it * 2 + (lane >> 4);
    int e  = lane & 15;
    int d  = idxs[wave][e];
    v8h qv = *(const v8h*)&q16[(size_t)d * DIM + hh * DHD];   // one 16B load
    const float* kr = &kf[e * HID + hh * DHD];
    float4 k0 = *(const float4*)kr;
    float4 k1 = *(const float4*)(kr + 4);
    float s = (float)qv[0] * k0.x + (float)qv[1] * k0.y + (float)qv[2] * k0.z + (float)qv[3] * k0.w
            + (float)qv[4] * k1.x + (float)qv[5] * k1.y + (float)qv[6] * k1.z + (float)qv[7] * k1.w;
    s *= inv_sqrt_dh;
    int eg = e0 + e; if (eg > E - 1) eg = E - 1;
    scores[(size_t)eg * NHEADS + hh] = s;
    atomicMaxF(&smax[(size_t)d * NHEADS + hh], s);
  }
}

__global__ void softmax_denom_kernel(const float* __restrict__ scores,
                                     const float* __restrict__ smax,
                                     const long long* __restrict__ ei,
                                     float* __restrict__ denom, int E) {
  int total = E * NHEADS;
  for (int i = blockIdx.x * blockDim.x + threadIdx.x; i < total; i += gridDim.x * blockDim.x) {
    int e = i >> 4, hh = i & 15;
    int d = (int)ei[(size_t)E + e];
    float ex = __expf(scores[i] - smax[d * NHEADS + hh]);
    atomicAdd(&denom[d * NHEADS + hh], ex);
  }
}

__global__ void scatter_out_kernel(const float* __restrict__ scores,
                                   const float* __restrict__ smax,
                                   const float* __restrict__ denom,
                                   const float* __restrict__ vsc,
                                   const float* __restrict__ rel_x,
                                   const long long* __restrict__ ei,
                                   float* __restrict__ out, int E) {
  for (int e = blockIdx.x * blockDim.x + threadIdx.x; e < E; e += gridDim.x * blockDim.x) {
    int d = (int)ei[(size_t)E + e];
    float s = 0.f;
    #pragma unroll
    for (int hh = 0; hh < NHEADS; ++hh) {
      float a = __expf(scores[(size_t)e * NHEADS + hh] - smax[d * NHEADS + hh])
                / denom[d * NHEADS + hh];
      s += a * vsc[(size_t)e * NHEADS + hh];
    }
    s *= (1.0f / (float)NHEADS);
    atomicAdd(&out[d * 3 + 0], s * rel_x[(size_t)e * 3 + 0]);
    atomicAdd(&out[d * 3 + 1], s * rel_x[(size_t)e * 3 + 1]);
    atomicAdd(&out[d * 3 + 2], s * rel_x[(size_t)e * 3 + 2]);
  }
}

// ---------------- launch ----------------

extern "C" void kernel_launch(void* const* d_in, const int* in_sizes, int n_in,
                              void* d_out, int out_size, void* d_ws, size_t ws_size,
                              hipStream_t stream) {
  const float*     h         = (const float*)d_in[0];
  const float*     rel_x     = (const float*)d_in[1];
  const float*     edge_feat = (const float*)d_in[2];
  const float*     e_w       = (const float*)d_in[3];
  const long long* ei        = (const long long*)d_in[4];   // [2,E]: row0=src, row1=dst
  const float* xk_w1 = (const float*)d_in[5];
  const float* xk_b1 = (const float*)d_in[6];
  const float* xk_g  = (const float*)d_in[7];
  const float* xk_be = (const float*)d_in[8];
  const float* xk_w2 = (const float*)d_in[9];
  const float* xk_b2 = (const float*)d_in[10];
  const float* xv_w1 = (const float*)d_in[11];
  const float* xv_b1 = (const float*)d_in[12];
  const float* xv_g  = (const float*)d_in[13];
  const float* xv_be = (const float*)d_in[14];
  const float* xv_w2 = (const float*)d_in[15];
  const float* xv_b2 = (const float*)d_in[16];
  const float* xq_w1 = (const float*)d_in[17];
  const float* xq_b1 = (const float*)d_in[18];
  const float* xq_g  = (const float*)d_in[19];
  const float* xq_be = (const float*)d_in[20];
  const float* xq_w2 = (const float*)d_in[21];
  const float* xq_b2 = (const float*)d_in[22];
  (void)n_in; (void)out_size; (void)ws_size;

  const int N = in_sizes[0] / DIM;
  const int E = in_sizes[1] / 3;

  char* wsb = (char*)d_ws;
  size_t off = 0;
  auto walloc = [&](size_t bytes) -> void* {
    void* p = wsb + off;
    off = (off + bytes + 255) & ~(size_t)255;
    return p;
  };
  _Float16* h16    = (_Float16*)walloc((size_t)N * DIM * sizeof(_Float16));
  _Float16* qbuf   = (_Float16*)walloc((size_t)N * DIM * sizeof(_Float16));
  float*    scores = (float*)walloc((size_t)E * NHEADS * sizeof(float));
  float*    vscb   = (float*)walloc((size_t)E * NHEADS * sizeof(float));
  float*    smax   = (float*)walloc((size_t)N * NHEADS * sizeof(float));
  float*    denom  = (float*)walloc((size_t)N * NHEADS * sizeof(float));
  _Float16* Pk1 = (_Float16*)walloc((size_t)9 * 8 * 512 * 2);
  _Float16* Pv1 = (_Float16*)walloc((size_t)9 * 8 * 512 * 2);
  _Float16* Pk2 = (_Float16*)walloc((size_t)4 * 8 * 512 * 2);
  _Float16* Pv2 = (_Float16*)walloc((size_t)4 * 1 * 512 * 2);
  _Float16* Pq1 = (_Float16*)walloc((size_t)4 * 8 * 512 * 2);
  _Float16* Pq2 = (_Float16*)walloc((size_t)4 * 8 * 512 * 2);

  float* outp = (float*)d_out;

  {
    int n16 = N * NHEADS, n3 = N * 3;
    int total = n16 > n3 ? n16 : n3;
    init_kernel<<<(total + 255) / 256, 256, 0, stream>>>(smax, denom, outp, n16, n3);
  }
  conv_h_kernel<<<((N * DIM / 4) + 255) / 256, 256, 0, stream>>>(h, h16, N * DIM / 4);
  pack_w_kernel<<<(9 * 8 * 512 + 255) / 256, 256, 0, stream>>>(xk_w1, Pk1, KVIN, 9, HID);
  pack_w_kernel<<<(9 * 8 * 512 + 255) / 256, 256, 0, stream>>>(xv_w1, Pv1, KVIN, 9, HID);
  pack_w_kernel<<<(4 * 8 * 512 + 255) / 256, 256, 0, stream>>>(xk_w2, Pk2, HID, 4, HID);
  pack_w_kernel<<<(4 * 1 * 512 + 255) / 256, 256, 0, stream>>>(xv_w2, Pv2, HID, 4, NHEADS);
  pack_w_kernel<<<(4 * 8 * 512 + 255) / 256, 256, 0, stream>>>(xq_w1, Pq1, DIM, 4, HID);
  pack_w_kernel<<<(4 * 8 * 512 + 255) / 256, 256, 0, stream>>>(xq_w2, Pq2, HID, 4, HID);

  const int Ntiles = (N + 15) / 16;
  qmlp_kernel<<<(Ntiles + 3) / 4, 128, 0, stream>>>(h16, Pq1, xq_b1, xq_g, xq_be,
                                                    Pq2, xq_b2, qbuf, N, Ntiles);

  const int Etiles = (E + 15) / 16;
  edge_kernel<<<(Etiles + 3) / 4, 128, 0, stream>>>(h16, edge_feat, e_w, ei, qbuf,
      Pk1, xk_b1, xk_g, xk_be, Pk2, xk_b2,
      Pv1, xv_b1, xv_g, xv_be, Pv2, xv_b2,
      scores, vscb, smax, E, Etiles);

  softmax_denom_kernel<<<((E * NHEADS) + 255) / 256, 256, 0, stream>>>(scores, smax, ei, denom, E);
  scatter_out_kernel<<<(E + 255) / 256, 256, 0, stream>>>(scores, smax, denom, vscb, rel_x, ei, outp, E);
}
